// EarthAttention3D_31911607009866
// MI455X (gfx1250) — compile-verified, matrix operand order checked by reading, not run on hardware
//
#include <hip/hip_runtime.h>
#include <stdint.h>

// ---------------- problem constants ----------------
#define NWIN  720
#define LTOK  144
#define DIMF  192
#define NHEAD 6
#define HD    32
#define NQKV  576           // 3*DIMF
#define TOW   240
#define SP    160           // padded score row (144 + 16 zero pad -> K multiple of 32)
#define MTOT  (NWIN*LTOK)   // 103680

typedef __attribute__((ext_vector_type(16))) _Float16 v16h;
typedef __attribute__((ext_vector_type(8)))  float    v8f;

struct alignas(16) U32x4 { uint32_t a, b, c, d; };
struct alignas(16) F32x4 { float x, y, z, w; };

__device__ __forceinline__ v8f wmma_16x16x32(v16h a, v16h b, v8f c) {
    // D = A(16x32 f16) * B(32x16 f16) + C(16x16 f32)
    return __builtin_amdgcn_wmma_f32_16x16x32_f16(
        /*neg_a=*/false, a, /*neg_b=*/false, b,
        /*c_mod=*/(short)0, c, /*reuse_a=*/false, /*reuse_b=*/false);
}

// A/B operand pattern (16-bit, wave32): lane L (row/col = L&15) holds
// K = kbase..kbase+7 in VGPR0..3 and K = kbase+16..kbase+23 in VGPR4..7,
// kbase = (L>>4)*8.  Caller passes p already offset by kbase; we load
// p[0..7] and p[16..23].
__device__ __forceinline__ v16h opnd_from_f32(const float* p) {
    const F32x4* q = (const F32x4*)p;
    F32x4 a0 = q[0], a1 = q[1], a2 = q[4], a3 = q[5];
    v16h r;
    r[0]=(_Float16)a0.x;  r[1]=(_Float16)a0.y;  r[2]=(_Float16)a0.z;  r[3]=(_Float16)a0.w;
    r[4]=(_Float16)a1.x;  r[5]=(_Float16)a1.y;  r[6]=(_Float16)a1.z;  r[7]=(_Float16)a1.w;
    r[8]=(_Float16)a2.x;  r[9]=(_Float16)a2.y;  r[10]=(_Float16)a2.z; r[11]=(_Float16)a2.w;
    r[12]=(_Float16)a3.x; r[13]=(_Float16)a3.y; r[14]=(_Float16)a3.z; r[15]=(_Float16)a3.w;
    return r;
}

__device__ __forceinline__ v16h opnd_from_f16(const _Float16* p) {
    union { v16h h; U32x4 u[2]; } r;
    r.u[0] = *(const U32x4*)(p);
    r.u[1] = *(const U32x4*)(p + 16);
    return r.h;
}

// Async global->LDS copy (CDNA5 GLOBAL_LOAD_ASYNC_TO_LDS_B128, ASYNCcnt-tracked).
// vdst = LDS byte offset (low 32 bits of the flat shared pointer), vaddr = 64-bit
// global address.  Per-lane transfer of 16 bytes, EXEC must cover all lanes used.
__device__ __forceinline__ void async_ld_b128(void* lds_ptr, const void* gptr) {
    uint32_t l = (uint32_t)(size_t)lds_ptr;   // flat shared addr[31:0] == LDS offset
    asm volatile("global_load_async_to_lds_b128 %0, %1, off"
                 :: "v"(l), "v"(gptr)
                 : "memory");
}
__device__ __forceinline__ void wait_async0() {
    asm volatile("s_wait_asynccnt 0" ::: "memory");
}

// =====================================================================
// Kernel A: qkv = x @ w1.T + b1   (M=103680, N=576, K=192)
// Output scattered to per-(window,head) contiguous f16 Q/K/V, Q pre-scaled.
// =====================================================================
__global__ __launch_bounds__(256) void qkv_gemm(
    const float* __restrict__ x, const float* __restrict__ w1,
    const float* __restrict__ b1,
    _Float16* __restrict__ qb, _Float16* __restrict__ kb,
    _Float16* __restrict__ vb)
{
    const int gw   = blockIdx.x * 8 + (threadIdx.x >> 5);
    const int lane = threadIdx.x & 31;
    const int nt = gw % (NQKV / 16);
    const int mt = gw / (NQKV / 16);
    const int m0 = mt * 16, n0 = nt * 16;
    const int r16 = lane & 15;
    const int hl  = lane >> 4;
    const int kb8 = hl * 8;

    const float* arow = x  + (size_t)(m0 + r16) * DIMF;   // token row (K contiguous)
    const float* brow = w1 + (size_t)(n0 + r16) * DIMF;   // w1 row = B column (K contiguous)

    v8f acc = {};
#pragma unroll
    for (int kt = 0; kt < DIMF / 32; ++kt) {
        v16h a = opnd_from_f32(arow + kt * 32 + kb8);
        v16h b = opnd_from_f32(brow + kt * 32 + kb8);
        acc = wmma_16x16x32(a, b, acc);
    }

    // D tile: lane holds column f = n0 + (lane&15); rows m0 + vv + 8*hl
    const int   f    = n0 + r16;
    const int   sect = f / DIMF;          // 0=q 1=k 2=v (tiles never straddle)
    const int   fo   = f % DIMF;
    const int   hh   = fo >> 5, dd = fo & 31;
    const float addb = b1[f];
    const float mul  = (sect == 0) ? 0.17677669529663687f : 1.0f;  // 1/sqrt(32)
    _Float16* dst = (sect == 0) ? qb : (sect == 1) ? kb : vb;

#pragma unroll
    for (int vv = 0; vv < 8; ++vv) {
        int m = m0 + vv + 8 * hl;
        int w = m / LTOK, t = m % LTOK;
        dst[(((size_t)w * NHEAD + hh) * LTOK + t) * HD + dd] =
            (_Float16)((acc[vv] + addb) * mul);
    }
}

// =====================================================================
// Kernel B: fused attention per (window, head).
// 288 threads = 9 waves; wave i owns the 16-row stripe [16i, 16i+16).
// =====================================================================
__global__ __launch_bounds__(288) void attn_kernel(
    const _Float16* __restrict__ qb, const _Float16* __restrict__ kb,
    const _Float16* __restrict__ vb, const float* __restrict__ mask,
    const float* __restrict__ bt,   _Float16* __restrict__ ob)
{
    __shared__ _Float16 qs[LTOK * HD];
    __shared__ _Float16 ks[LTOK * HD];
    __shared__ _Float16 vs[LTOK * HD];
    __shared__ _Float16 vt[HD * SP];      // transposed V, K-padded with zeros
    __shared__ float    Ssm[LTOK * SP];   // scores then probs (f32)

    const int wh = blockIdx.x;            // 0..4319
    const int w  = wh / NHEAD;
    const int h  = wh % NHEAD;
    const int tid  = threadIdx.x;
    const int wave = tid >> 5;
    const int lane = tid & 31;

    // ---- async global->LDS staging of q/k/v (3 x 9216 B, 576 b128 each) ----
    {
        const char* gq = (const char*)(qb + (size_t)wh * LTOK * HD);
        const char* gk = (const char*)(kb + (size_t)wh * LTOK * HD);
        const char* gv = (const char*)(vb + (size_t)wh * LTOK * HD);
        char* lq = (char*)qs;
        char* lk = (char*)ks;
        char* lv = (char*)vs;
        // 576 transfers per array, 288 threads -> exactly 2 iterations, EXEC full
        for (int i = tid; i < (LTOK * HD * 2) / 16; i += 288) {
            async_ld_b128(lq + i * 16, gq + i * 16);
            async_ld_b128(lk + i * 16, gk + i * 16);
            async_ld_b128(lv + i * 16, gv + i * 16);
        }
        wait_async0();
    }
    __syncthreads();

    const int it   = wave;        // i-tile (query stripe)
    const int r16  = lane & 15;
    const int hl   = lane >> 4;
    const int kb8  = hl * 8;

    // ---- stage 1: S = (q*scale) @ k^T, K=32 -> one WMMA per 16x16 tile ----
    v16h aop = opnd_from_f16(qs + (it * 16 + r16) * HD + kb8);
#pragma unroll
    for (int jt = 0; jt < 9; ++jt) {
        v16h bop = opnd_from_f16(ks + (jt * 16 + r16) * HD + kb8);
        v8f acc = {};
        acc = wmma_16x16x32(aop, bop, acc);
        int n = jt * 16 + r16;
#pragma unroll
        for (int vv = 0; vv < 8; ++vv) {
            int m = it * 16 + vv + 8 * hl;
            Ssm[m * SP + n] = acc[vv];
        }
    }
    // zero the K-pad columns (144..159) of this wave's rows
    for (int e = lane; e < 256; e += 32)
        Ssm[(it * 16 + (e >> 4)) * SP + LTOK + (e & 15)] = 0.0f;

    // ---- build transposed V with zero K-padding: vt[d][t], t in [0,160) ----
    for (int i = tid; i < HD * SP; i += 288) {
        int d = i / SP, t = i % SP;
        vt[i] = (t < LTOK) ? vs[t * HD + d] : (_Float16)0.0f;
    }

    // ---- bias + mask + softmax (wave-local: lanes l and l^16 split a row) ----
    {
        const int r  = it * 16 + r16;
        const int zi = r / 72, hi2 = (r / 12) % 6, wi = r % 12;
        const float* mrow  = mask + ((size_t)w * LTOK + r) * LTOK;
        const int    tow6h = (w % TOW) * NHEAD + h;
        const int    j0    = hl * 72;
        float* srow = Ssm + r * SP;

        float mx = -3.4e38f;
        for (int j = j0; j < j0 + 72; ++j) {
            int zj = j / 72, hj = (j / 12) % 6, wj = j % 12;
            int idx = (zi + 2 * zj) * 828 + (hi2 + 6 * hj) * 23 + (wi - wj + 11);
            float val = srow[j] + bt[(size_t)idx * (TOW * NHEAD) + tow6h] + mrow[j];
            srow[j] = val;
            mx = fmaxf(mx, val);
        }
        mx = fmaxf(mx, __shfl_xor(mx, 16, 32));
        float sum = 0.0f;
        for (int j = j0; j < j0 + 72; ++j) {
            float e = __expf(srow[j] - mx);
            srow[j] = e;
            sum += e;
        }
        sum += __shfl_xor(sum, 16, 32);
        float inv = 1.0f / sum;
        for (int j = j0; j < j0 + 72; ++j) srow[j] *= inv;
    }
    __syncthreads();   // vt (cross-thread) must be complete before PV

    // ---- stage 2: O = P @ V  (K padded to 160 -> 5 clean WMMA steps) ----
#pragma unroll
    for (int nt2 = 0; nt2 < 2; ++nt2) {
        v8f acc = {};
        const int nn = nt2 * 16 + r16;
#pragma unroll
        for (int kt = 0; kt < 5; ++kt) {
            v16h a = opnd_from_f32(Ssm + (it * 16 + r16) * SP + kt * 32 + kb8);
            v16h b = opnd_from_f16(vt + nn * SP + kt * 32 + kb8);
            acc = wmma_16x16x32(a, b, acc);
        }
        const int dcol = h * HD + nt2 * 16 + r16;
#pragma unroll
        for (int vv = 0; vv < 8; ++vv) {
            int t = it * 16 + vv + 8 * hl;
            ob[((size_t)w * LTOK + t) * DIMF + dcol] = (_Float16)acc[vv];
        }
    }
}

// =====================================================================
// Kernel C: out = O @ w2.T + b2   (M=103680, N=192, K=192), f32 output
// =====================================================================
__global__ __launch_bounds__(256) void proj_gemm(
    const _Float16* __restrict__ ob, const float* __restrict__ w2,
    const float* __restrict__ b2, float* __restrict__ out)
{
    const int gw   = blockIdx.x * 8 + (threadIdx.x >> 5);
    const int lane = threadIdx.x & 31;
    const int nt = gw % (DIMF / 16);
    const int mt = gw / (DIMF / 16);
    const int m0 = mt * 16, n0 = nt * 16;
    const int r16 = lane & 15;
    const int hl  = lane >> 4;
    const int kb8 = hl * 8;

    const _Float16* arow = ob + (size_t)(m0 + r16) * DIMF;
    const float*    brow = w2 + (size_t)(n0 + r16) * DIMF;

    v8f acc = {};
#pragma unroll
    for (int kt = 0; kt < DIMF / 32; ++kt) {
        v16h a = opnd_from_f16(arow + kt * 32 + kb8);
        v16h b = opnd_from_f32(brow + kt * 32 + kb8);
        acc = wmma_16x16x32(a, b, acc);
    }

    const float addb = b2[n0 + r16];
#pragma unroll
    for (int vv = 0; vv < 8; ++vv) {
        int m = m0 + vv + 8 * hl;
        out[(size_t)m * DIMF + n0 + r16] = acc[vv] + addb;
    }
}

// =====================================================================
// Host launcher
// =====================================================================
extern "C" void kernel_launch(void* const* d_in, const int* in_sizes, int n_in,
                              void* d_out, int out_size, void* d_ws, size_t ws_size,
                              hipStream_t stream) {
    (void)in_sizes; (void)n_in; (void)out_size; (void)ws_size;
    const float* x    = (const float*)d_in[0];
    const float* mask = (const float*)d_in[1];
    const float* w1   = (const float*)d_in[2];
    const float* b1   = (const float*)d_in[3];
    const float* w2   = (const float*)d_in[4];
    const float* b2   = (const float*)d_in[5];
    const float* bt   = (const float*)d_in[6];
    float* out = (float*)d_out;

    const size_t qkv_e = (size_t)NWIN * NHEAD * LTOK * HD;  // 19,906,560 halves
    _Float16* qb = (_Float16*)d_ws;
    _Float16* kb = qb + qkv_e;
    _Float16* vb = kb + qkv_e;
    _Float16* ob = vb + qkv_e;                              // NWIN*LTOK*DIMF halves

    // A: (103680/16)*(576/16) = 233280 wave-tiles, 8 waves/block
    qkv_gemm<<<233280 / 8, 256, 0, stream>>>(x, w1, b1, qb, kb, vb);
    // B: one block per (window, head)
    attn_kernel<<<NWIN * NHEAD, 288, 0, stream>>>(qb, kb, vb, mask, bt, ob);
    // C: (103680/16)*(192/16) = 77760 wave-tiles
    proj_gemm<<<77760 / 8, 256, 0, stream>>>(ob, w2, b2, out);
}
